// WindowAttentionGlobal_3135326126800
// MI455X (gfx1250) — matrix operand order repaired and checked by reading, not compiled, not run on hardware
//
#include <hip/hip_runtime.h>
#include <hip/hip_bf16.h>

typedef _Float16 v16h __attribute__((ext_vector_type(16)));
typedef _Float16 v8h  __attribute__((ext_vector_type(8)));
typedef _Float16 v4h  __attribute__((ext_vector_type(4)));
typedef float    v8f  __attribute__((ext_vector_type(8)));

#define DIM   128
#define HEADS 4
#define HD    32
#define NTOK  49
#define NPAD  64
#define QK_SCALE 0.17677669529663687f   // 1/sqrt(32)

// ---------------- fragment helpers (layouts per CDNA5 ISA 7.12.2) ----------------

__device__ __forceinline__ v16h frag_cat(v8h lo, v8h hi) {
  return __builtin_shufflevector(lo, hi, 0,1,2,3,4,5,6,7,8,9,10,11,12,13,14,15);
}

// A-matrix 16x32 f16 fragment from row-major storage.
// lane<16: K in {k0..k0+7} u {k0+16..k0+23}; lane>=16: +8 offset.
__device__ __forceinline__ v16h load_a(const _Float16* base, int ld, int r0, int k0, int lane) {
  int row = r0 + (lane & 15);
  int kk  = k0 + ((lane & 31) < 16 ? 0 : 8);
  const _Float16* p = base + row * ld + kk;
  v8h lo = *(const v8h*)(p);
  v8h hi = *(const v8h*)(p + 16);
  return frag_cat(lo, hi);
}

// B-matrix 32x16 f16 fragment from "n-major" storage S[n][k] (contiguous in k).
// lane = column N; lane<16: K in {k0..k0+7} u {k0+16..k0+23}; lane>=16: +8 offset.
__device__ __forceinline__ v16h load_b(const _Float16* base, int ld, int n0, int k0, int lane) {
  int n  = n0 + (lane & 15);
  int kk = k0 + ((lane & 31) < 16 ? 0 : 8);
  const _Float16* p = base + n * ld + kk;
  v8h lo = *(const v8h*)(p);
  v8h hi = *(const v8h*)(p + 16);
  return frag_cat(lo, hi);
}

__device__ __forceinline__ v8f wmma_f16(v16h a, v16h b, v8f c) {
  return __builtin_amdgcn_wmma_f32_16x16x32_f16(false, a, false, b, (short)0, c, false, false);
}

__device__ __forceinline__ v8f splat8(float v) {
  v8f r = {v, v, v, v, v, v, v, v};
  return r;
}

// ---------------- prep kernel: f16 weight transposes, scaled q, padded bias ----------------

__global__ void prep_kernel(const float* __restrict__ qg, const float* __restrict__ tbl,
                            const float* __restrict__ wkv, const float* __restrict__ wpj,
                            const int* __restrict__ ridx,
                            _Float16* __restrict__ wkv_t, _Float16* __restrict__ wpj_t,
                            _Float16* __restrict__ qf, float* __restrict__ biasf) {
  int i = blockIdx.x * 256 + threadIdx.x;
  const int SZ_WKV = 2 * DIM * DIM;         // 32768  : wkv_t[c][k], c<256,k<128
  const int SZ_WPJ = DIM * DIM;             // 16384  : wpj_t[c][k]
  const int SZ_Q   = 64 * HEADS * NPAD * HD;// 524288 : qf[img][h][row(64)][d]
  const int SZ_B   = HEADS * NPAD * NPAD;   // 16384  : biasf[h][n][m]
  if (i < SZ_WKV) {
    int c = i >> 7, k = i & 127;
    wkv_t[i] = (_Float16)wkv[k * 256 + c];
  } else if ((i -= SZ_WKV) < SZ_WPJ) {
    int c = i >> 7, k = i & 127;
    wpj_t[i] = (_Float16)wpj[k * 128 + c];
  } else if ((i -= SZ_WPJ) < SZ_Q) {
    int d = i & 31, row = (i >> 5) & 63, h = (i >> 11) & 3, im = i >> 13;
    float v = 0.f;
    if (row < NTOK) v = qg[(((im * HEADS) + h) * NTOK + row) * HD + d] * QK_SCALE;
    qf[i] = (_Float16)v;
  } else if ((i -= SZ_Q) < SZ_B) {
    int m = i & 63, n = (i >> 6) & 63, h = i >> 12;
    float v;
    if (m >= NTOK)      v = -1e30f;               // mask padded keys through softmax
    else if (n >= NTOK) v = 0.f;                  // padded query rows: benign
    else                v = tbl[ridx[n * NTOK + m] * HEADS + h];
    biasf[i] = v;
  }
}

// ---------------- fused window-attention kernel: one workgroup per window ----------------

__global__ __launch_bounds__(256) void win_attn_kernel(
    const float* __restrict__ x, const float* __restrict__ b_kv,
    const float* __restrict__ b_proj,
    const _Float16* __restrict__ wkv_t, const _Float16* __restrict__ wpj_t,
    const _Float16* __restrict__ qf, const float* __restrict__ biasf,
    float* __restrict__ out) {
  __shared__ _Float16 xs[NPAD * DIM];          // 16KB: x tile f16; reused as attn-out staging
  __shared__ _Float16 Ks[NPAD * DIM];          // 16KB: K  [token][h*32+d]
  __shared__ _Float16 Vt[DIM * NPAD];          // 16KB: V^T [h*32+d][token]
  __shared__ _Float16 Ps[HEADS * NPAD * NPAD]; // 32KB: softmax probs per head

  const int tid  = threadIdx.x;
  const int lane = tid & 31;
  const int wv   = tid >> 5;                   // 8 waves
  const int hi8  = (lane < 16) ? 0 : 8;
  const int b    = blockIdx.x;
  const int img  = b >> 6;

  // ---- Phase 1: x -> xs (f16), rows 49..63 zero-padded ----
  {
    const float4* x4 = (const float4*)(x + (size_t)b * NTOK * DIM);
    for (int i = tid; i < NPAD * (DIM / 4); i += 256) {
      int row = i >> 5, c4 = i & 31;
      float4 v = make_float4(0.f, 0.f, 0.f, 0.f);
      if (row < NTOK) v = x4[row * 32 + c4];
      v4h hv; hv[0] = (_Float16)v.x; hv[1] = (_Float16)v.y;
              hv[2] = (_Float16)v.z; hv[3] = (_Float16)v.w;
      *(v4h*)&xs[row * DIM + c4 * 4] = hv;
    }
  }
  __syncthreads();

  // ---- Phase 2: KV = xs @ Wkv + b_kv ; K -> Ks, V -> Vt (transposed store) ----
  // Bias folded into the WMMA C operand: D-fragment column is lane&15 for all 8
  // rows, so a bias-splat accumulator applies b_kv[col] with zero extra VALU.
#pragma unroll
  for (int t = 0; t < 2; ++t) {
    int nt  = wv * 2 + t;
    int cb  = nt * 16;
    int col = cb + (lane & 15);
    v8f cinit = splat8(b_kv[col]);
    v16h Bf[4];
#pragma unroll
    for (int kt = 0; kt < 4; ++kt) Bf[kt] = load_b(wkv_t, DIM, cb, kt * 32, lane);
#pragma unroll
    for (int mt = 0; mt < 4; ++mt) {
      v8f acc = cinit;
#pragma unroll
      for (int kt = 0; kt < 4; ++kt) {
        v16h Af = load_a(xs, DIM, mt * 16, kt * 32, lane);
        acc = wmma_f16(Af, Bf[kt], acc);
      }
      int rbase = mt * 16 + hi8;
      if (cb < DIM) {                            // K half (channels 0..127)
#pragma unroll
        for (int r = 0; r < 8; ++r)
          Ks[(rbase + r) * DIM + col] = (_Float16)acc[r];
      } else {                                   // V half: per-lane 8 rows are contiguous
        int vc = col - DIM;
        v8h pk;
#pragma unroll
        for (int r = 0; r < 8; ++r) pk[r] = (_Float16)acc[r];
        *(v8h*)&Vt[vc * NPAD + rbase] = pk;
      }
    }
  }
  __syncthreads();

  // ---- Phase 3: S = (q*scale) @ K^T + bias ; softmax ; P -> Ps ----
  const int h  = wv >> 1;    // head
  const int mh = wv & 1;     // row half (rows mh*32 .. mh*32+31)
  v8f S[2][4];
  {
    v16h Qf[2], Kf[4];
    const _Float16* qbase = qf + (size_t)(img * HEADS + h) * NPAD * HD;
#pragma unroll
    for (int i = 0; i < 2; ++i) Qf[i] = load_a(qbase, HD, (2 * mh + i) * 16, 0, lane);
#pragma unroll
    for (int nt = 0; nt < 4; ++nt) Kf[nt] = load_b(Ks, DIM, nt * 16, h * HD, lane);
    // Relative-position bias loaded straight into the C fragment (no post-add).
    const float* bh = biasf + h * NPAD * NPAD;
#pragma unroll
    for (int i = 0; i < 2; ++i) {
      int rbase = (2 * mh + i) * 16 + hi8;
#pragma unroll
      for (int nt = 0; nt < 4; ++nt) {
        int colb = nt * 16 + (lane & 15);
        v8f bf;
#pragma unroll
        for (int r = 0; r < 8; ++r) bf[r] = bh[(rbase + r) * NPAD + colb];
        S[i][nt] = wmma_f16(Qf[i], Kf[nt], bf);  // K = head_dim = 32: one WMMA per tile
      }
    }
    // softmax over 64 columns: 4 tiles in-register + 16-lane xor reduction
#pragma unroll
    for (int i = 0; i < 2; ++i) {
      int rbase = (2 * mh + i) * 16 + hi8;
#pragma unroll
      for (int r = 0; r < 8; ++r) {
        float mx = fmaxf(fmaxf(S[i][0][r], S[i][1][r]), fmaxf(S[i][2][r], S[i][3][r]));
#pragma unroll
        for (int msk = 1; msk <= 8; msk <<= 1) mx = fmaxf(mx, __shfl_xor(mx, msk, 32));
        float sum = 0.f;
#pragma unroll
        for (int nt = 0; nt < 4; ++nt) {
          float e = __expf(S[i][nt][r] - mx);
          S[i][nt][r] = e;
          sum += e;
        }
#pragma unroll
        for (int msk = 1; msk <= 8; msk <<= 1) sum += __shfl_xor(sum, msk, 32);
        float inv = 1.0f / sum;
#pragma unroll
        for (int nt = 0; nt < 4; ++nt)
          Ps[(h * NPAD + rbase + r) * NPAD + nt * 16 + (lane & 15)] =
              (_Float16)(S[i][nt][r] * inv);
      }
    }
  }

  // ---- Phase 4: O = P @ V  (no barrier: wave consumes only its own P rows) ----
  {
    v8f O[2][2] = {{{}, {}}, {{}, {}}};
    v16h Vf[2][2];
#pragma unroll
    for (int nt = 0; nt < 2; ++nt)
#pragma unroll
      for (int kb = 0; kb < 2; ++kb)
        Vf[nt][kb] = load_b(Vt, NPAD, h * HD + nt * 16, kb * 32, lane);
    const _Float16* pb = Ps + h * NPAD * NPAD;
#pragma unroll
    for (int i = 0; i < 2; ++i)
#pragma unroll
      for (int kb = 0; kb < 2; ++kb) {
        v16h Af = load_a(pb, NPAD, (2 * mh + i) * 16, kb * 32, lane);
#pragma unroll
        for (int nt = 0; nt < 2; ++nt) O[i][nt] = wmma_f16(Af, Vf[nt][kb], O[i][nt]);
      }
    // stage attention output (channel = h*32+d) into xs (x tile is dead now)
#pragma unroll
    for (int i = 0; i < 2; ++i) {
      int rbase = (2 * mh + i) * 16 + hi8;
#pragma unroll
      for (int nt = 0; nt < 2; ++nt) {
        int colb = h * HD + nt * 16 + (lane & 15);
#pragma unroll
        for (int r = 0; r < 8; ++r)
          xs[(rbase + r) * DIM + colb] = (_Float16)O[i][nt][r];
      }
    }
  }
  __syncthreads();

  // ---- Phase 5: out = O @ Wproj + b_proj (bias via C operand; fp32 stores) ----
  {
    int cb = wv * 16;
    int col = cb + (lane & 15);
    v8f cinit = splat8(b_proj[col]);
    v16h Wf[4];
#pragma unroll
    for (int kt = 0; kt < 4; ++kt) Wf[kt] = load_b(wpj_t, DIM, cb, kt * 32, lane);
#pragma unroll
    for (int mt = 0; mt < 4; ++mt) {
      v8f acc = cinit;
#pragma unroll
      for (int kt = 0; kt < 4; ++kt) {
        v16h Af = load_a(xs, DIM, mt * 16, kt * 32, lane);
        acc = wmma_f16(Af, Wf[kt], acc);
      }
      int rbase = mt * 16 + hi8;
#pragma unroll
      for (int r = 0; r < 8; ++r) {
        int row = rbase + r;
        if (row < NTOK)
          out[((size_t)b * NTOK + row) * DIM + col] = acc[r];
      }
    }
  }
}

// ---------------- launch ----------------

extern "C" void kernel_launch(void* const* d_in, const int* in_sizes, int n_in,
                              void* d_out, int out_size, void* d_ws, size_t ws_size,
                              hipStream_t stream) {
  const float* x    = (const float*)d_in[0];
  const float* qg   = (const float*)d_in[1];
  const float* tbl  = (const float*)d_in[2];
  const float* wkv  = (const float*)d_in[3];
  const float* bkv  = (const float*)d_in[4];
  const float* wpj  = (const float*)d_in[5];
  const float* bpj  = (const float*)d_in[6];
  const int*   ridx = (const int*)d_in[7];
  float* out = (float*)d_out;

  char* ws = (char*)d_ws;
  _Float16* wkv_t = (_Float16*)(ws);                            // 64 KB
  _Float16* wpj_t = (_Float16*)(ws + 65536);                    // 32 KB
  _Float16* qf    = (_Float16*)(ws + 65536 + 32768);            // 1 MB
  float*    biasf = (float*)(ws + 65536 + 32768 + 1048576);     // 64 KB

  const int total = 32768 + 16384 + 524288 + 16384;             // 589824 = 2304*256
  prep_kernel<<<total / 256, 256, 0, stream>>>(qg, tbl, wkv, wpj, ridx,
                                               wkv_t, wpj_t, qf, biasf);
  win_attn_kernel<<<4096, 256, 0, stream>>>(x, bkv, bpj, wkv_t, wpj_t, qf, biasf, out);
}
